// GAT_net_86930138071075
// MI455X (gfx1250) — compile-verified, matrix operand order checked by reading, not run on hardware
//
#include <hip/hip_runtime.h>
#include <hip/hip_bf16.h>

typedef __attribute__((ext_vector_type(16))) __bf16 v16bf;
typedef __attribute__((ext_vector_type(8)))  float  v8f;
typedef __attribute__((ext_vector_type(2)))  float  v2f;

#define H128 128

// ---------- helpers ----------
__device__ __forceinline__ unsigned f2ord(float f) {
  unsigned u = __float_as_uint(f);
  return (u & 0x80000000u) ? ~u : (u | 0x80000000u);
}
__device__ __forceinline__ float ord2f(unsigned u) {
  unsigned v = (u & 0x80000000u) ? (u & 0x7FFFFFFFu) : ~u;
  return __uint_as_float(v);
}

// ---------- GEMM layer 1: [N,4] x [4,128] via V_WMMA_F32_16X16X4_F32 ----------
__global__ void gemm_l1(const float* __restrict__ X, const float* __restrict__ W,
                        float* __restrict__ Hout, int nTiles) {
  int tid  = threadIdx.x;
  int wave = (blockIdx.x * blockDim.x + tid) >> 5;
  if (wave >= nTiles) return;
  int lane = tid & 31, half = lane >> 4, l15 = lane & 15;

  // A fragment: 16x4 f32. lanes 0-15: K=0,1 ; lanes 16-31: K=2,3
  const float* xr = X + (size_t)(wave * 16 + l15) * 4 + 2 * half;
  v2f a; a[0] = xr[0]; a[1] = xr[1];

  v8f c[8] = {};
#if __has_builtin(__builtin_amdgcn_wmma_f32_16x16x4_f32)
#pragma unroll
  for (int t = 0; t < 8; t++) {
    int n = t * 16 + l15;
    v2f b; b[0] = W[(2 * half) * H128 + n]; b[1] = W[(2 * half + 1) * H128 + n];
    c[t] = __builtin_amdgcn_wmma_f32_16x16x4_f32(false, a, false, b,
                                                 (short)0, c[t], false, false);
  }
#else
  // VALU fallback (should not trigger on gfx1250)
#pragma unroll
  for (int t = 0; t < 8; t++) {
    int n = t * 16 + l15;
    for (int r = 0; r < 8; r++) {
      float acc = 0.f;
      const float* row = X + (size_t)(wave * 16 + r + 8 * half) * 4;
      for (int k = 0; k < 4; k++) acc += row[k] * W[k * H128 + n];
      c[t][r] = acc;
    }
  }
#endif
  float* hr = Hout + (size_t)(wave * 16) * H128;
#pragma unroll
  for (int t = 0; t < 8; t++)
#pragma unroll
    for (int r = 0; r < 8; r++)
      hr[(size_t)(r + 8 * half) * H128 + t * 16 + l15] = c[t][r];
}

// ---------- GEMM layers 2/3: [N,128] x [128,128] via bf16 WMMA ----------
// W staged in LDS pre-packed into the exact B-fragment register layout so each
// fragment read is two ds_load_b128.
__global__ void gemm_bf16(const float* __restrict__ X, const float* __restrict__ W,
                          float* __restrict__ Hout, int nTiles) {
  __shared__ __attribute__((aligned(32))) __bf16 wlds[16384];
  int tid = threadIdx.x;
  for (int idx = tid; idx < 16384; idx += 256) {
    int j  = idx & 15;
    int ln = (idx >> 4) & 31;
    int t  = (idx >> 9) & 7;
    int ks = idx >> 12;
    int hf = ln >> 4;
    int n  = t * 16 + (ln & 15);
    int kk = ks * 32;
    int k  = (j < 8) ? (kk + 8 * hf + j) : (kk + 8 + 8 * hf + j);
    wlds[idx] = (__bf16)W[k * H128 + n];
  }
  __syncthreads();

  int wave = (blockIdx.x * blockDim.x + tid) >> 5;
  if (wave >= nTiles) return;
  int lane = tid & 31, half = lane >> 4, l15 = lane & 15;

  const float* xr = X + (size_t)(wave * 16 + l15) * H128;
  v8f c[8] = {};

#pragma unroll
  for (int ks = 0; ks < 4; ks++) {
    int kk = ks * 32;
    v16bf a;
    const float* p0 = xr + kk + 8 * half;        // K = kk+8h .. +7
    const float* p1 = xr + kk + 16 + 8 * half;   // K = kk+16+8h .. +7
#pragma unroll
    for (int j = 0; j < 8; j++) { a[j] = (__bf16)p0[j]; a[j + 8] = (__bf16)p1[j]; }
#pragma unroll
    for (int t = 0; t < 8; t++) {
      v16bf b = *(const v16bf*)(&wlds[((ks * 8 + t) * 32 + lane) * 16]);
      c[t] = __builtin_amdgcn_wmma_f32_16x16x32_bf16(false, a, false, b,
                                                     (short)0, c[t], false, false);
    }
  }
  float* hr = Hout + (size_t)(wave * 16) * H128;
#pragma unroll
  for (int t = 0; t < 8; t++)
#pragma unroll
    for (int r = 0; r < 8; r++)
      hr[(size_t)(r + 8 * half) * H128 + t * 16 + l15] = c[t][r];
}

// ---------- per-node attention logits: als = h.a_src, ald = h.a_dst ----------
__global__ void rowdot(const float* __restrict__ Hh, const float* __restrict__ as_,
                       const float* __restrict__ ad_, float* __restrict__ als,
                       float* __restrict__ ald, int N) {
  int tid  = blockIdx.x * blockDim.x + threadIdx.x;
  int node = tid >> 5, lane = tid & 31;
  if (node >= N) return;
  float4 h = *(const float4*)(Hh + (size_t)node * H128 + lane * 4);
  float4 a = *(const float4*)(as_ + lane * 4);
  float4 b = *(const float4*)(ad_ + lane * 4);
  float s = h.x * a.x + h.y * a.y + h.z * a.z + h.w * a.w;
  float t = h.x * b.x + h.y * b.y + h.z * b.z + h.w * b.w;
  for (int o = 16; o; o >>= 1) { s += __shfl_xor(s, o, 32); t += __shfl_xor(t, o, 32); }
  if (lane == 0) { als[node] = s; ald[node] = t; }
}

// ---------- init / zero ----------
__global__ void init_seg(unsigned* __restrict__ mord, float* __restrict__ den, int N) {
  int i = blockIdx.x * blockDim.x + threadIdx.x;
  if (i < N) { mord[i] = f2ord(-3.402823466e+38f); den[i] = 0.f; }
}
__global__ void zero_f32(float* __restrict__ p, long n) {
  long i = (long)blockIdx.x * blockDim.x + threadIdx.x;
  if (i < n) p[i] = 0.f;
}

// ---------- edge pass 1: leaky-relu score + segment max (ordered-uint atomicMax) ----------
__global__ void edge_max(const int* __restrict__ ei, const float* __restrict__ als,
                         const float* __restrict__ ald, float* __restrict__ esc,
                         unsigned* __restrict__ mord, int E0, int E) {
  int e = blockIdx.x * blockDim.x + threadIdx.x;
  if (e >= E) return;
  int s, d;
  if (e < E0) { s = ei[e]; d = ei[E0 + e]; } else { s = d = e - E0; }
  float sc = als[s] + ald[d];
  sc = sc > 0.f ? sc : 0.2f * sc;
  esc[e] = sc;
  atomicMax(mord + d, f2ord(sc));
}

// ---------- edge pass 2: exp(e - m[dst]) + segment sum ----------
__global__ void edge_exp(const int* __restrict__ ei, float* __restrict__ esc,
                         const unsigned* __restrict__ mord, float* __restrict__ den,
                         int E0, int E) {
  int e = blockIdx.x * blockDim.x + threadIdx.x;
  if (e >= E) return;
  int d  = (e < E0) ? ei[E0 + e] : (e - E0);
  float ex = __expf(esc[e] - ord2f(mord[d]));
  esc[e] = ex;
  atomicAdd(den + d, ex);
}

// ---------- edge pass 3: out[dst] += alpha * h[src]  (wave per edge) ----------
__global__ void edge_agg(const int* __restrict__ ei, const float* __restrict__ esc,
                         const float* __restrict__ den, const float* __restrict__ Hh,
                         float* __restrict__ out, int E0, int E) {
  int tid = blockIdx.x * blockDim.x + threadIdx.x;
  int e = tid >> 5;
  if (e >= E) return;
  int lane = tid & 31;
  int s, d;
  if (e < E0) { s = ei[e]; d = ei[E0 + e]; } else { s = d = e - E0; }
  float alpha = esc[e] / (den[d] + 1e-16f);
  float4 h = *(const float4*)(Hh + (size_t)s * H128 + lane * 4);
  float* op = out + (size_t)d * H128 + lane * 4;
  atomicAdd(op + 0, alpha * h.x);
  atomicAdd(op + 1, alpha * h.y);
  atomicAdd(op + 2, alpha * h.z);
  atomicAdd(op + 3, alpha * h.w);
}

// ---------- bias (+ optional relu) ----------
__global__ void bias_act(float* __restrict__ out, const float* __restrict__ b,
                         int N, int doRelu) {
  int i = blockIdx.x * blockDim.x + threadIdx.x;
  if (i >= N * H128) return;
  float v = out[i] + b[i & (H128 - 1)];
  if (doRelu) v = v > 0.f ? v : 0.f;
  out[i] = v;
}

// ---------- global mean pool (sums + counts) ----------
__global__ void pool_sum(const float* __restrict__ Hh, const int* __restrict__ batch,
                         float* __restrict__ pool, float* __restrict__ cnt, int N) {
  int tid  = blockIdx.x * blockDim.x + threadIdx.x;
  int node = tid >> 5, lane = tid & 31;
  if (node >= N) return;
  int g = batch[node];
  float4 h = *(const float4*)(Hh + (size_t)node * H128 + lane * 4);
  float* pp = pool + (size_t)g * H128 + lane * 4;
  atomicAdd(pp + 0, h.x);
  atomicAdd(pp + 1, h.y);
  atomicAdd(pp + 2, h.z);
  atomicAdd(pp + 3, h.w);
  if (lane == 0) atomicAdd(cnt + g, 1.0f);
}

// ---------- final linear: [64,128] x [128,2] + bias ----------
__global__ void final_lin(const float* __restrict__ pool, const float* __restrict__ cnt,
                          const float* __restrict__ Wl, const float* __restrict__ bl,
                          float* __restrict__ out) {
  int tid = threadIdx.x;     // 128 threads: g = tid>>1, col = tid&1
  int g = tid >> 1, cc = tid & 1;
  float cv = cnt[g]; cv = cv > 1.f ? cv : 1.f;
  float inv = 1.f / cv;
  float acc = bl[cc];
  for (int k = 0; k < H128; k++) acc += pool[g * H128 + k] * inv * Wl[k * 2 + cc];
  out[g * 2 + cc] = acc;
}

extern "C" void kernel_launch(void* const* d_in, const int* in_sizes, int n_in,
                              void* d_out, int out_size, void* d_ws, size_t ws_size,
                              hipStream_t stream) {
  const float* x     = (const float*)d_in[0];
  const int*   ei    = (const int*)d_in[1];
  const int*   batch = (const int*)d_in[2];
  const float* W1 = (const float*)d_in[3];
  const float* as1 = (const float*)d_in[4];
  const float* ad1 = (const float*)d_in[5];
  const float* b1  = (const float*)d_in[6];
  const float* W2 = (const float*)d_in[7];
  const float* as2 = (const float*)d_in[8];
  const float* ad2 = (const float*)d_in[9];
  const float* b2  = (const float*)d_in[10];
  const float* W3 = (const float*)d_in[11];
  const float* as3 = (const float*)d_in[12];
  const float* ad3 = (const float*)d_in[13];
  const float* b3  = (const float*)d_in[14];
  const float* Wlin = (const float*)d_in[15];
  const float* blin = (const float*)d_in[16];
  float* outp = (float*)d_out;

  const int N  = in_sizes[2];        // 50000 nodes (batch length)
  const int E0 = in_sizes[1] / 2;    // 800000 edges
  const int E  = E0 + N;             // + self loops
  const int G  = 64;

  // workspace layout (all f32/u32)
  float*    bufA  = (float*)d_ws;                 // [N,128] h (post-GEMM)
  float*    bufB  = bufA + (size_t)N * H128;      // [N,128] aggregated / next input
  float*    als   = bufB + (size_t)N * H128;      // [N]
  float*    ald   = als + N;                      // [N]
  unsigned* mord  = (unsigned*)(ald + N);         // [N]
  float*    den   = (float*)(mord + N);           // [N]
  float*    esc   = den + N;                      // [E]
  float*    pool  = esc + E;                      // [64,128]
  float*    cnt   = pool + (size_t)G * H128;      // [64]

  const int nTiles   = (N + 15) / 16;                    // 3125 (N divisible by 16)
  const int gemmBlks = (nTiles * 32 + 255) / 256;
  const int nodeWaveBlks = (N * 32 + 255) / 256;
  const int edgeBlks = (E + 255) / 256;
  const int edgeWaveBlks = ((long)E * 32 + 255) / 256;
  const int featBlks = ((long)N * H128 + 255) / 256;

  const float* Ws[3]  = {W1, W2, W3};
  const float* aSs[3] = {as1, as2, as3};
  const float* aDs[3] = {ad1, ad2, ad3};
  const float* bs[3]  = {b1, b2, b3};

  const float* Xin = x;
  for (int L = 0; L < 3; L++) {
    if (L == 0)
      gemm_l1<<<gemmBlks, 256, 0, stream>>>(Xin, Ws[0], bufA, nTiles);
    else
      gemm_bf16<<<gemmBlks, 256, 0, stream>>>(Xin, Ws[L], bufA, nTiles);

    rowdot<<<nodeWaveBlks, 256, 0, stream>>>(bufA, aSs[L], aDs[L], als, ald, N);
    init_seg<<<(N + 255) / 256, 256, 0, stream>>>(mord, den, N);
    zero_f32<<<featBlks, 256, 0, stream>>>(bufB, (long)N * H128);

    edge_max<<<edgeBlks, 256, 0, stream>>>(ei, als, ald, esc, mord, E0, E);
    edge_exp<<<edgeBlks, 256, 0, stream>>>(ei, esc, mord, den, E0, E);
    edge_agg<<<edgeWaveBlks, 256, 0, stream>>>(ei, esc, den, bufA, bufB, E0, E);

    bias_act<<<featBlks, 256, 0, stream>>>(bufB, bs[L], N, (L < 2) ? 1 : 0);
    Xin = bufB;
  }

  zero_f32<<<(G * H128 + G + 255) / 256, 256, 0, stream>>>(pool, (long)G * H128 + G);
  pool_sum<<<nodeWaveBlks, 256, 0, stream>>>(bufB, batch, pool, cnt, N);
  final_lin<<<1, 128, 0, stream>>>(pool, cnt, Wlin, blin, outp);
}
